// MeshConv_12360915878142
// MI455X (gfx1250) — compile-verified, matrix operand order checked by reading, not run on hardware
//
#include <hip/hip_runtime.h>

// MeshConv (ChebNet K=6) for MI455X / gfx1250, wave32.
// B=8, M=40000, FIN=64, K=6, FOUT=64, NNZ=320000.
// Chebyshev buffers kept in input layout [B][M][FIN] so x == T0 and each
// per-k GEMM slice is a contiguous row-major [B*M, 64] matrix.
// GEMM fuses two Chebyshev orders per pass (K=128) to halve out RMW traffic;
// A tiles staged into LDS with GLOBAL_LOAD_ASYNC_TO_LDS_B128 (ASYNCcnt path).

typedef __attribute__((ext_vector_type(16))) _Float16 v16h;
typedef __attribute__((ext_vector_type(8)))  float    v8f;

#define MROWS 40000
#define NNZE  320000
#define NB    8
#define NFIN  64
#define NCHEB 6
#define NFOUT 64

// ---------------- CSR build ----------------

__global__ __launch_bounds__(256) void zero_ints(int* p, int n) {
    int i = blockIdx.x * 256 + threadIdx.x;
    if (i < n) p[i] = 0;
}

__global__ __launch_bounds__(256) void edge_hist(const int* __restrict__ rows,
                                                 int* __restrict__ hist, int nnz) {
    int i = blockIdx.x * 256 + threadIdx.x;
    if (i < nnz) atomicAdd(&hist[rows[i]], 1);
}

// Single-block exclusive scan over M=40000 histogram bins (cheap, one launch).
__global__ __launch_bounds__(1024) void scan_offsets(const int* __restrict__ hist,
                                                     int* __restrict__ offs,
                                                     int* __restrict__ cursor, int M) {
    __shared__ int sh[1024];
    __shared__ int carry;
    int t = threadIdx.x;
    if (t == 0) carry = 0;
    __syncthreads();
    for (int base = 0; base < M; base += 1024) {
        int v = (base + t < M) ? hist[base + t] : 0;
        sh[t] = v;
        __syncthreads();
        for (int off = 1; off < 1024; off <<= 1) {   // Hillis-Steele inclusive scan
            int add = (t >= off) ? sh[t - off] : 0;
            __syncthreads();
            sh[t] += add;
            __syncthreads();
        }
        int excl = carry + sh[t] - v;
        if (base + t < M) { offs[base + t] = excl; cursor[base + t] = excl; }
        __syncthreads();
        if (t == 0) carry += sh[1023];
        __syncthreads();
    }
    if (t == 0) offs[M] = carry;
}

__global__ __launch_bounds__(256) void edge_scatter(const int* __restrict__ rows,
                                                    const int* __restrict__ cols,
                                                    const float* __restrict__ vals,
                                                    int* __restrict__ cursor,
                                                    int* __restrict__ ccol,
                                                    float* __restrict__ cval, int nnz) {
    int i = blockIdx.x * 256 + threadIdx.x;
    if (i < nnz) {
        int r = rows[i];
        int p = atomicAdd(&cursor[r], 1);
        ccol[p] = cols[i];
        cval[p] = vals[i];
    }
}

// ---------------- Chebyshev SpMM step ----------------
// Tout[:, m, :] = a * ( sum_e w_e * Tin[:, col_e, :] - Tin[:, m, :] ) - b * Tprev[:, m, :]
// One block per row m; 128 threads x float4 covers 8*64 = 512 floats.
// Tin (82 MB) is L2-resident (192 MB L2) so the random gathers stay on-chip.
__global__ __launch_bounds__(128)
void spmm_cheb(const float* __restrict__ Tin, const float* __restrict__ Tprev,
               float* __restrict__ Tout, const int* __restrict__ offs,
               const int* __restrict__ ccol, const float* __restrict__ cval,
               float a, float b) {
    const int m  = blockIdx.x;
    const int t  = threadIdx.x;
    const int bb = t >> 4;
    const int fq = (t & 15) << 2;
    const size_t plane = (size_t)MROWS * NFIN;
    const size_t base  = (size_t)bb * plane + fq;

    const int s = offs[m], e = offs[m + 1];

    float4 acc = *(const float4*)(Tin + base + (size_t)m * NFIN);
    acc.x = -acc.x; acc.y = -acc.y; acc.z = -acc.z; acc.w = -acc.w;

    for (int i = s; i < e; ++i) {
        const int   c = ccol[i];
        const float w = cval[i];
        float4 v = *(const float4*)(Tin + base + (size_t)c * NFIN);
        acc.x += w * v.x; acc.y += w * v.y; acc.z += w * v.z; acc.w += w * v.w;
    }

    float4 r;
    if (b != 0.f) {
        float4 p = *(const float4*)(Tprev + base + (size_t)m * NFIN);
        r.x = a * acc.x - b * p.x; r.y = a * acc.y - b * p.y;
        r.z = a * acc.z - b * p.z; r.w = a * acc.w - b * p.w;
    } else {
        r.x = a * acc.x; r.y = a * acc.y; r.z = a * acc.z; r.w = a * acc.w;
    }
    *(float4*)(Tout + base + (size_t)m * NFIN) = r;
}

// ---------------- WMMA GEMM pass (two Chebyshev orders ka, kb) ----------------
// out[R, o] (+)= sum_f Ta[R, f] * W[f*6+ka, o] + sum_f Tb[R, f] * W[f*6+kb, o]
// Block = 256 threads = 8 wave32s; each wave owns a 16-row tile (block = 128 rows).
// A tiles staged f32 into LDS via GLOBAL_LOAD_ASYNC_TO_LDS_B128 (stride 68 floats
// = 272 B rows, 16 B aligned); f32->f16 happens at operand pack. W slices staged
// transposed f16 [o][f]. 4 K-chunks x 4 N-tiles -> 16 v_wmma_f32_16x16x32_f16.
#define LDSTRIDE 68

__global__ __launch_bounds__(256)
void gemm_pass2(const float* __restrict__ Ta, const float* __restrict__ Tb,
                const float* __restrict__ W, float* __restrict__ out,
                int ka, int kb, int accum) {
    __shared__ float    Atf[8][2][16 * LDSTRIDE];   // per-wave f32 tiles, 69.6 KB
    __shared__ _Float16 WhT[2][NFOUT * LDSTRIDE];   // [src][o][f] f16, 17.4 KB

    const int  t    = threadIdx.x;
    const int  lane = t & 31;
    const int  wv   = t >> 5;
    const long rowBase = (long)blockIdx.x * 128;

    // --- async-stage this wave's two 16x64 f32 A tiles straight into LDS ---
    {
        const int rsel = lane >> 4;          // row parity within a 2-row / 512 B copy
        const int cb   = (lane & 15) * 16;   // column byte 0..240
        for (int src = 0; src < 2; ++src) {
            const float* Tsrc  = src ? Tb : Ta;
            const float* gbase = Tsrc + (rowBase + (long)wv * 16) * NFIN;
            #pragma unroll
            for (int rr = 0; rr < 16; rr += 2) {
                const int r = rr + rsel;
                unsigned ldsaddr =
                    (unsigned)(uintptr_t)(&Atf[wv][src][r * LDSTRIDE]) + (unsigned)cb;
                const char* g = (const char*)(gbase + (long)r * NFIN) + cb;
                asm volatile("global_load_async_to_lds_b128 %0, %1, off"
                             :: "v"(ldsaddr), "v"(g) : "memory");
            }
        }
    }

    // --- stage transposed f16 W slices for ka, kb: WhT[src][o][f] = W[f*6+k][o] ---
    for (int i = t; i < 2 * NFOUT * NFIN; i += 256) {
        const int src = i >> 12;
        const int r   = i & 4095;
        const int o = r >> 6, f = r & 63;
        const int k = src ? kb : ka;
        WhT[src][o * LDSTRIDE + f] = (_Float16)W[(f * NCHEB + k) * NFOUT + o];
    }

    asm volatile("s_wait_asynccnt 0x0" ::: "memory");   // A tiles landed in LDS
    __syncthreads();                                    // WhT visible block-wide

    const int lr = lane & 15;   // A: row-in-tile, B: column n
    const int h  = lane >> 4;   // K-half select per CDNA5 16-bit operand layout

    v8f acc[4];
    for (int nt = 0; nt < 4; ++nt) {
        const int o = nt * 16 + lr;
        #pragma unroll
        for (int i = 0; i < 8; ++i) {
            const long R = rowBase + (long)wv * 16 + i + 8 * h;  // C/D: VGPR i -> M = i + 8h
            acc[nt][i] = accum ? out[R * NFOUT + o] : 0.f;
        }
    }

    #pragma unroll
    for (int src = 0; src < 2; ++src) {
        #pragma unroll
        for (int chunk = 0; chunk < 2; ++chunk) {
            // A operand: element pair (2j,2j+1) -> K = (j<4 ? 8h+2j : 16+8h+2(j-4)) + {0,1}
            v16h a;
            #pragma unroll
            for (int j = 0; j < 8; ++j) {
                const int K = (j < 4) ? (8 * h + 2 * j) : (16 + 8 * h + 2 * (j - 4));
                const int f = chunk * 32 + K;
                a[2 * j]     = (_Float16)Atf[wv][src][lr * LDSTRIDE + f];
                a[2 * j + 1] = (_Float16)Atf[wv][src][lr * LDSTRIDE + f + 1];
            }
            #pragma unroll
            for (int nt = 0; nt < 4; ++nt) {
                const int o = nt * 16 + lr;
                v16h bm;
                #pragma unroll
                for (int j = 0; j < 8; ++j) {
                    const int K = (j < 4) ? (8 * h + 2 * j) : (16 + 8 * h + 2 * (j - 4));
                    const int f = chunk * 32 + K;
                    bm[2 * j]     = WhT[src][o * LDSTRIDE + f];
                    bm[2 * j + 1] = WhT[src][o * LDSTRIDE + f + 1];
                }
                acc[nt] = __builtin_amdgcn_wmma_f32_16x16x32_f16(
                    /*neg_a=*/false, a, /*neg_b=*/false, bm,
                    /*c_mod=*/(short)0, acc[nt], /*reuse_a=*/false, /*reuse_b=*/false);
            }
        }
    }

    for (int nt = 0; nt < 4; ++nt) {
        const int o = nt * 16 + lr;
        #pragma unroll
        for (int i = 0; i < 8; ++i) {
            const long R = rowBase + (long)wv * 16 + i + 8 * h;
            out[R * NFOUT + o] = acc[nt][i];
        }
    }
}

// ---------------- host-side launch ----------------

extern "C" void kernel_launch(void* const* d_in, const int* in_sizes, int n_in,
                              void* d_out, int out_size, void* d_ws, size_t ws_size,
                              hipStream_t stream) {
    (void)in_sizes; (void)n_in; (void)out_size; (void)ws_size;

    const float* x    = (const float*)d_in[0];   // [B, M, FIN] f32  (== T0 in our layout)
    const float* ev   = (const float*)d_in[1];   // [NNZ] f32
    const float* W    = (const float*)d_in[2];   // [FIN*K, FOUT] f32
    const int*   erow = (const int*)d_in[3];     // [NNZ] i32
    const int*   ecol = (const int*)d_in[4];     // [NNZ] i32
    float* out = (float*)d_out;                  // [B, M, FOUT] f32

    const size_t planeBytes = (size_t)NB * MROWS * NFIN * sizeof(float);  // 81.92 MB

    char*  ws  = (char*)d_ws;
    size_t off = 0;
    auto alloc = [&](size_t bytes) -> void* {
        off = (off + 255) & ~(size_t)255;
        void* p = ws + off;
        off += bytes;
        return p;
    };
    float* bufA   = (float*)alloc(planeBytes);
    float* bufB   = (float*)alloc(planeBytes);
    float* bufC   = (float*)alloc(planeBytes);
    int*   hist   = (int*)alloc(MROWS * sizeof(int));
    int*   cursor = (int*)alloc(MROWS * sizeof(int));
    int*   offs   = (int*)alloc((MROWS + 1) * sizeof(int));
    int*   ccol   = (int*)alloc(NNZE * sizeof(int));
    float* cval   = (float*)alloc(NNZE * sizeof(float));

    // CSR build (rebuilt every call; no cached state)
    zero_ints<<<(MROWS + 255) / 256, 256, 0, stream>>>(hist, MROWS);
    edge_hist<<<(NNZE + 255) / 256, 256, 0, stream>>>(erow, hist, NNZE);
    scan_offsets<<<1, 1024, 0, stream>>>(hist, offs, cursor, MROWS);
    edge_scatter<<<(NNZE + 255) / 256, 256, 0, stream>>>(erow, ecol, ev, cursor, ccol, cval, NNZE);

    const int gemmGrid = (NB * MROWS) / 128;   // 2500 blocks, 128 rows each, exact cover

    // T1 = S(x) = L@x - x
    spmm_cheb<<<MROWS, 128, 0, stream>>>(x, x, bufA, offs, ccol, cval, 1.f, 0.f);
    // pass (k0,k1): initializes out
    gemm_pass2<<<gemmGrid, 256, 0, stream>>>(x, bufA, W, out, 0, 1, 0);
    // T2 = 2 S(T1) - T0 ; T3 = 2 S(T2) - T1
    spmm_cheb<<<MROWS, 128, 0, stream>>>(bufA, x, bufB, offs, ccol, cval, 2.f, 1.f);
    spmm_cheb<<<MROWS, 128, 0, stream>>>(bufB, bufA, bufC, offs, ccol, cval, 2.f, 1.f);
    // pass (k2,k3): accumulate
    gemm_pass2<<<gemmGrid, 256, 0, stream>>>(bufB, bufC, W, out, 2, 3, 1);
    // T4 = 2 S(T3) - T2  (bufA reusable: T1 consumed) ; T5 = 2 S(T4) - T3 (bufB reusable)
    spmm_cheb<<<MROWS, 128, 0, stream>>>(bufC, bufB, bufA, offs, ccol, cval, 2.f, 1.f);
    spmm_cheb<<<MROWS, 128, 0, stream>>>(bufA, bufC, bufB, offs, ccol, cval, 2.f, 1.f);
    // pass (k4,k5): accumulate
    gemm_pass2<<<gemmGrid, 256, 0, stream>>>(bufA, bufB, W, out, 4, 5, 1);
}
